// TRANSFORMER_CRF_NER_77103252898536
// MI455X (gfx1250) — matrix operand order repaired and checked by reading, not compiled, not run on hardware
//
#include <hip/hip_runtime.h>
#include <hip/hip_bf16.h>
#include <stdint.h>

#define NUM_B 64
#define SEQ_T 512
#define HID   768
#define NLAB  128
#define NEGV  -10000.0f
#define MTOT  (NUM_B * SEQ_T)   // 32768

typedef __attribute__((ext_vector_type(16))) __bf16 v16bf;
typedef __attribute__((ext_vector_type(8)))  float  v8f;

union FragU { unsigned u[8]; v16bf v; };

__device__ __forceinline__ unsigned pack_bf16(float lo, float hi) {
  // round-to-nearest-even f32 -> bf16, packed pair
  unsigned a = __float_as_uint(lo);
  unsigned b = __float_as_uint(hi);
  a += 0x7FFFu + ((a >> 16) & 1u);
  b += 0x7FFFu + ((b >> 16) & 1u);
  return (a >> 16) | (b & 0xFFFF0000u);
}

// ---------------------------------------------------------------------------
// Kernel 0: one-time W (128x768 f32) -> packed bf16 pairs (128x384 dwords)
// ---------------------------------------------------------------------------
__global__ __launch_bounds__(256) void convert_w(
    const float* __restrict__ W, unsigned* __restrict__ WB)
{
  int p = blockIdx.x * 256 + threadIdx.x;      // 49152 pairs total
  float2 v = *(const float2*)(W + 2 * p);
  WB[p] = pack_bf16(v.x, v.y);
}

// ---------------------------------------------------------------------------
// Kernel 1: feats[m][n] = sum_k seq_out[m][k] * W[n][k] + bias[n]
// Block = 8 waves sharing one 16-row A strip staged once in LDS as bf16.
// Each wave owns one 16-col N tile; K-loop of v_wmma_f32_16x16x32_bf16.
// ---------------------------------------------------------------------------
#define ARS 388   // LDS A row stride in dwords (384 pairs + 4 pad, kills bank conflicts)

__global__ __launch_bounds__(256) void gemm_feats(
    const float* __restrict__ A,      // seq_out, (MTOT x HID)
    const unsigned* __restrict__ WB,  // bf16-packed W, (NLAB x 384 dwords)
    const float* __restrict__ bias,   // (NLAB)
    float* __restrict__ feats)        // (MTOT x NLAB)
{
  __shared__ unsigned a_lds[16 * ARS];          // 24.8 KB
  const int tid = threadIdx.x;
  const int m_tile = blockIdx.x;                // 2048 strips

  // Cooperative stage + convert: 16 x 768 f32 -> 6144 packed-bf16 dwords
  const float* abase = A + (size_t)m_tile * 16 * HID;
  #pragma unroll
  for (int i = 0; i < 24; ++i) {
    int p   = tid + 256 * i;                    // pair index 0..6143
    int row = p / 384;
    int cp  = p - row * 384;
    float2 v = *(const float2*)(abase + row * HID + 2 * cp);
    a_lds[row * ARS + cp] = pack_bf16(v.x, v.y);
  }
  __syncthreads();

  const int lane = tid & 31;
  const int wave = tid >> 5;                    // n_tile 0..7
  const int l15  = lane & 15;
  const int half = lane >> 4;
  const int n    = wave * 16 + l15;

  const unsigned* arow = a_lds + l15 * ARS;
  const unsigned* brow = WB + (size_t)n * 384;

  v8f acc = {};
  #pragma unroll 8
  for (int k0 = 0; k0 < HID; k0 += 32) {
    const int kd = k0 >> 1;                     // pair index base
    // A fragment: v[0..3] K=8h+2j ; v[4..7] K=16+8h+2j   (ds_load_b128 x2)
    uint4 a0 = *(const uint4*)(arow + kd + 4 * half);
    uint4 a1 = *(const uint4*)(arow + kd + 8 + 4 * half);
    // B fragment: v[0..3] K=8h+2j ; v[4..7] K=16+8h+2j  (global_load_b128 x2, L2-resident W)
    uint4 b0 = *(const uint4*)(brow + kd + 4 * half);
    uint4 b1 = *(const uint4*)(brow + kd + 8 + 4 * half);
    FragU a, b;
    a.u[0] = a0.x; a.u[1] = a0.y; a.u[2] = a0.z; a.u[3] = a0.w;
    a.u[4] = a1.x; a.u[5] = a1.y; a.u[6] = a1.z; a.u[7] = a1.w;
    b.u[0] = b0.x; b.u[1] = b0.y; b.u[2] = b0.z; b.u[3] = b0.w;
    b.u[4] = b1.x; b.u[5] = b1.y; b.u[6] = b1.z; b.u[7] = b1.w;
    acc = __builtin_amdgcn_wmma_f32_16x16x32_bf16(false, a.v, false, b.v,
                                                  (short)0, acc, false, false);
  }
  // C/D layout: lanes 0-15 N=lane, VGPR r -> M=r; lanes 16-31 -> M=8+r
  const float bn = bias[n];
  #pragma unroll
  for (int r = 0; r < 8; ++r) {
    int m = m_tile * 16 + half * 8 + r;
    feats[(size_t)m * NLAB + n] = acc[r] + bn;
  }
}

// ---------------------------------------------------------------------------
// Kernel 2: per-batch CRF forward (logsumexp) + Viterbi scan + gold score.
// One block per batch element, one thread per label. transitions cached in
// LDS (stride 129 => conflict-free column reads). Next-step feats row is
// prefetched with global_load_async_to_lds_b32 (ASYNCcnt) into a double
// buffer while the current step's 128-wide reductions run.
// ---------------------------------------------------------------------------
__global__ __launch_bounds__(128) void crf_scan(
    const float* __restrict__ feats,   // (B*T x L)
    const float* __restrict__ trans,   // (L x L)
    const int*   __restrict__ labels,  // (B x T)
    uint8_t*     __restrict__ psi,     // (T-1, B, L)
    float*       __restrict__ fwd_s,   // (B)
    float*       __restrict__ gold_s,  // (B)
    int*         __restrict__ last_s,  // (B)
    float*       __restrict__ out_best)// d_out + 1 : best_score (B)
{
  __shared__ float trans_s[NLAB * 129];
  __shared__ float alpha[NLAB];
  __shared__ float delta[NLAB];
  __shared__ float red[NLAB];
  __shared__ int   redi[NLAB];
  __shared__ float fbuf[2][NLAB];

  const int b = blockIdx.x;
  const int j = threadIdx.x;

  for (int idx = j; idx < NLAB * NLAB; idx += 128) {
    int r = idx >> 7, c = idx & 127;
    trans_s[r * 129 + c] = trans[idx];
  }
  alpha[j] = (j == 0) ? 0.0f : NEGV;   // START == 0
  delta[j] = (j == 0) ? 0.0f : NEGV;
  __syncthreads();

  const float* frow = feats + (size_t)b * SEQ_T * NLAB;
  const float* trow = &trans_s[j * 129];

  // async prefetch of feats row t=1 into fbuf[1]
  {
    unsigned lds0 = (unsigned)(unsigned long long)&fbuf[1][j];
    const float* gp = frow + (size_t)1 * NLAB + j;
    asm volatile("global_load_async_to_lds_b32 %0, %1, off"
                 :: "v"(lds0), "v"(gp) : "memory");
  }

  for (int t = 1; t < SEQ_T; ++t) {
    asm volatile("s_wait_asynccnt 0x0" ::: "memory");
    float f = fbuf[t & 1][j];
    // issue prefetch for next step while we compute
    int tn = (t + 1 < SEQ_T) ? (t + 1) : (SEQ_T - 1);
    {
      unsigned ldsn = (unsigned)(unsigned long long)&fbuf[tn & 1][j];
      const float* gp = frow + (size_t)tn * NLAB + j;
      asm volatile("global_load_async_to_lds_b32 %0, %1, off"
                   :: "v"(ldsn), "v"(gp) : "memory");
    }

    float ma = -3.4e38f, md = -3.4e38f;
    int   arg = 0;
    #pragma unroll 4
    for (int i = 0; i < NLAB; ++i) {
      float tr = trow[i];
      float va = tr + alpha[i];
      float vd = tr + delta[i];
      ma = fmaxf(ma, va);
      if (vd > md) { md = vd; arg = i; }
    }
    float s = 0.0f;
    #pragma unroll 4
    for (int i = 0; i < NLAB; ++i)
      s += __expf(trow[i] + alpha[i] - ma);
    float na = ma + __logf(s) + f;
    float nd = md + f;
    psi[((size_t)(t - 1) * NUM_B + b) * NLAB + j] = (uint8_t)arg;
    __syncthreads();                 // all reads of alpha/delta done
    alpha[j] = na;
    delta[j] = nd;
    __syncthreads();                 // writes visible for next step
  }

  // forward_score = logsumexp(alpha)
  red[j] = alpha[j]; __syncthreads();
  for (int off = 64; off > 0; off >>= 1) {
    if (j < off) red[j] = fmaxf(red[j], red[j + off]);
    __syncthreads();
  }
  float amax = red[0]; __syncthreads();
  red[j] = __expf(alpha[j] - amax); __syncthreads();
  for (int off = 64; off > 0; off >>= 1) {
    if (j < off) red[j] += red[j + off];
    __syncthreads();
  }
  float fscore = amax + __logf(red[0]); __syncthreads();

  // best_score = max(delta), last = argmax (tree reduction, ties keep the lower slot)
  red[j] = delta[j]; redi[j] = j; __syncthreads();
  for (int off = 64; off > 0; off >>= 1) {
    if (j < off && red[j + off] > red[j]) { red[j] = red[j + off]; redi[j] = redi[j + off]; }
    __syncthreads();
  }
  float best = red[0]; int last = redi[0]; __syncthreads();

  // gold score: sum_t trans[lt,lp] + feats[b,t,lt], t = 1..T-1
  const int* lab = labels + (size_t)b * SEQ_T;
  float g = 0.0f;
  for (int t = 1 + j; t < SEQ_T; t += 128) {
    int lt = lab[t], lp = lab[t - 1];
    g += trans_s[lt * 129 + lp] + frow[(size_t)t * NLAB + lt];
  }
  red[j] = g; __syncthreads();
  for (int off = 64; off > 0; off >>= 1) {
    if (j < off) red[j] += red[j + off];
    __syncthreads();
  }
  if (j == 0) {
    fwd_s[b]  = fscore;
    gold_s[b] = red[0];
    last_s[b] = last;
    out_best[b] = best;
  }
}

// ---------------------------------------------------------------------------
// Kernel 3: backtrace (one thread per batch) + loss = mean(fwd - gold)
// ---------------------------------------------------------------------------
__global__ __launch_bounds__(64) void backtrace_loss(
    const uint8_t* __restrict__ psi,
    const int*   __restrict__ last_s,
    const float* __restrict__ fwd_s,
    const float* __restrict__ gold_s,
    float*       __restrict__ out)   // d_out: [loss(1), best(64), path(64*512)]
{
  __shared__ float red[64];
  const int b = threadIdx.x;
  float* path = out + 1 + NUM_B;
  int cur = last_s[b];
  path[(size_t)b * SEQ_T + (SEQ_T - 1)] = (float)cur;
  for (int t = SEQ_T - 2; t >= 0; --t) {
    cur = psi[((size_t)t * NUM_B + b) * NLAB + cur];
    path[(size_t)b * SEQ_T + t] = (float)cur;
  }
  red[b] = fwd_s[b] - gold_s[b];
  __syncthreads();
  for (int off = 32; off > 0; off >>= 1) {
    if (b < off) red[b] += red[b + off];
    __syncthreads();
  }
  if (b == 0) out[0] = red[0] / (float)NUM_B;
}

// ---------------------------------------------------------------------------
extern "C" void kernel_launch(void* const* d_in, const int* in_sizes, int n_in,
                              void* d_out, int out_size, void* d_ws, size_t ws_size,
                              hipStream_t stream) {
  const float* seq_out = (const float*)d_in[0];
  const float* W       = (const float*)d_in[1];
  const float* bias    = (const float*)d_in[2];
  const float* trans   = (const float*)d_in[3];
  const int*   labels  = (const int*)d_in[4];
  float* out = (float*)d_out;

  char* ws = (char*)d_ws;
  float* feats = (float*)ws;                                   // 16 MB
  size_t off = (size_t)MTOT * NLAB * sizeof(float);
  unsigned* wb = (unsigned*)(ws + off);                        // 192 KB bf16 W
  off += (size_t)NLAB * (HID / 2) * sizeof(unsigned);
  uint8_t* psi = (uint8_t*)(ws + off);                         // 4.19 MB
  off += (size_t)(SEQ_T - 1) * NUM_B * NLAB;
  off = (off + 255) & ~(size_t)255;
  float* fwd_s  = (float*)(ws + off); off += NUM_B * sizeof(float);
  float* gold_s = (float*)(ws + off); off += NUM_B * sizeof(float);
  int*   last_s = (int*)(ws + off);   off += NUM_B * sizeof(int);

  convert_w<<<(NLAB * HID / 2) / 256, 256, 0, stream>>>(W, wb);
  gemm_feats<<<MTOT / 16, 256, 0, stream>>>(seq_out, wb, bias, feats);
  crf_scan<<<NUM_B, 128, 0, stream>>>(feats, trans, labels, psi,
                                      fwd_s, gold_s, last_s, out + 1);
  backtrace_loss<<<1, 64, 0, stream>>>(psi, last_s, fwd_s, gold_s, out);
}
